// SelfScaledDotProductAttention2D_55241869361691
// MI455X (gfx1250) — matrix operand order repaired and checked by reading, not compiled, not run on hardware
//

#include <hip/hip_runtime.h>
#include <hip/hip_bf16.h>

typedef __attribute__((ext_vector_type(16))) _Float16 v16h;
typedef __attribute__((ext_vector_type(8)))  _Float16 v8h;
typedef __attribute__((ext_vector_type(8)))  float    v8f;

#define NB   8
#define CCH  256
#define HWS  4096   // 64*64 spatial positions

static __device__ __forceinline__ v16h cat8(v8h lo, v8h hi) {
    return __builtin_shufflevector(lo, hi, 0,1,2,3,4,5,6,7,8,9,10,11,12,13,14,15);
}

static __device__ __forceinline__ v8f wmma_f16(v16h a, v16h b, v8f c) {
    // D = A(16x32 f16) * B(32x16 f16) + C(16x16 f32)
    return __builtin_amdgcn_wmma_f32_16x16x32_f16(false, a, false, b, (short)0, c, false, false);
}

// ---------------------------------------------------------------------------
// Kernel 0: convert fp32 weights to fp16 (row-major [out][in]; contiguous K
// per output row == exactly the wave32 WMMA B-fragment access pattern).
// ---------------------------------------------------------------------------
__global__ void cvt_w_kernel(const float* __restrict__ wq, const float* __restrict__ wk,
                             const float* __restrict__ wv,
                             _Float16* __restrict__ oq, _Float16* __restrict__ ok,
                             _Float16* __restrict__ ov) {
    int i = blockIdx.x * blockDim.x + threadIdx.x;
    if (i < CCH * CCH) {
        oq[i] = (_Float16)wq[i];
        ok[i] = (_Float16)wk[i];
        ov[i] = (_Float16)wv[i];
    }
}

// Double-buffered 256-deep K accumulation of one 16x16 output tile.
static __device__ __forceinline__ v8f proj_tile(const v16h (&af)[8],
                                                const _Float16* __restrict__ w,
                                                int o0, int lo, int hi) {
    const _Float16* wrow = w + (size_t)(o0 + lo) * CCH + hi * 16;
    v8f acc = {};
    v16h b_next = *(const v16h*)wrow;
#pragma unroll
    for (int c0 = 0; c0 < 8; ++c0) {
        v16h b = b_next;
        if (c0 < 7) b_next = *(const v16h*)(wrow + (c0 + 1) * 32);
        acc = wmma_f16(af[c0], b, acc);
    }
    return acc;
}

// ---------------------------------------------------------------------------
// Kernel 1: LayerNorm(channel) + Q/K/V projection via WMMA.
// One wave handles a 16-position tile. q pre-scaled by 1/sqrt(C). v stored
// transposed [n][c][t] so the PV GEMM's B-fragment is a contiguous load.
// ---------------------------------------------------------------------------
__global__ __launch_bounds__(128) void lnqkv_kernel(
        const float* __restrict__ x, const float* __restrict__ gamma,
        const float* __restrict__ beta,
        const _Float16* __restrict__ w16q, const float* __restrict__ bq,
        const _Float16* __restrict__ w16k, const float* __restrict__ bk,
        const _Float16* __restrict__ w16v, const float* __restrict__ bv,
        _Float16* __restrict__ q16, _Float16* __restrict__ k16,
        _Float16* __restrict__ vt16) {
    __shared__ _Float16 xln[4][16][CCH];   // 32 KB

    const int wave = threadIdx.x >> 5;
    const int lane = threadIdx.x & 31;
    const int tile = blockIdx.x * 4 + wave;          // 2048 tiles total
    const int n    = tile / (HWS / 16);
    const int sp0  = (tile % (HWS / 16)) * 16;
    const int lo   = lane & 15;
    const int hi   = lane >> 4;
    const int cbase = hi * 128;                      // lane pair splits channels

    // ---- pass 1: load x (strided by HW per channel), accumulate stats ----
    const float* xb = x + (size_t)n * CCH * HWS + (size_t)cbase * HWS + sp0 + lo;
    float s = 0.f, ss = 0.f;
#pragma unroll 8
    for (int i = 0; i < 128; ++i) {
        float v = xb[(size_t)i * HWS];
        s += v; ss += v * v;
        xln[wave][lo][cbase + i] = (_Float16)v;
    }
    s  += __shfl_xor(s, 16);
    ss += __shfl_xor(ss, 16);
    const float mean = s * (1.0f / CCH);
    const float var  = ss * (1.0f / CCH) - mean * mean;
    const float rstd = rsqrtf(var + 1e-5f);

    // ---- pass 2: normalize in LDS (fp16) ----
#pragma unroll 8
    for (int i = 0; i < 128; ++i) {
        int c = cbase + i;
        float xv = (float)xln[wave][lo][c];
        xln[wave][lo][c] = (_Float16)((xv - mean) * rstd * gamma[c] + beta[c]);
    }
    asm volatile("s_wait_dscnt 0x0" ::: "memory");

    // ---- A-fragments of x_ln (16x32 fp16, contiguous k = 16*hi + s) ----
    v16h af[8];
#pragma unroll
    for (int c0 = 0; c0 < 8; ++c0) {
        const _Float16* p = &xln[wave][lo][c0 * 32 + hi * 16];
        af[c0] = cat8(*(const v8h*)p, *(const v8h*)(p + 8));
    }

    // ---- Q ----
    for (int o0 = 0; o0 < CCH; o0 += 16) {
        v8f acc = proj_tile(af, w16q, o0, lo, hi);
        const float bias = bq[o0 + lo];
#pragma unroll
        for (int r = 0; r < 8; ++r) {
            int rr = r + hi * 8;
            q16[((size_t)n * HWS + sp0 + rr) * CCH + o0 + lo] =
                (_Float16)((acc[r] + bias) * 0.0625f);   // fold 1/sqrt(C)
        }
    }
    // ---- K ----
    for (int o0 = 0; o0 < CCH; o0 += 16) {
        v8f acc = proj_tile(af, w16k, o0, lo, hi);
        const float bias = bk[o0 + lo];
#pragma unroll
        for (int r = 0; r < 8; ++r) {
            int rr = r + hi * 8;
            k16[((size_t)n * HWS + sp0 + rr) * CCH + o0 + lo] =
                (_Float16)(acc[r] + bias);
        }
    }
    // ---- V (transposed [c][t]) ----
    for (int o0 = 0; o0 < CCH; o0 += 16) {
        v8f acc = proj_tile(af, w16v, o0, lo, hi);
        const float bias = bv[o0 + lo];
#pragma unroll
        for (int r = 0; r < 8; ++r) {
            int rr = r + hi * 8;
            vt16[((size_t)n * CCH + o0 + lo) * HWS + sp0 + rr] =
                (_Float16)(acc[r] + bias);
        }
    }
}

// ---------------------------------------------------------------------------
// Kernel 2: flash attention. One wave owns a 16-query tile; loops over the
// 4096 keys in chunks of 32 with online softmax. 32 WMMA per chunk, with
// double-buffered fragment loads so VMEM latency hides under the XDL ops.
// ---------------------------------------------------------------------------
__global__ __launch_bounds__(128) void attn_kernel(
        const _Float16* __restrict__ q16, const _Float16* __restrict__ k16,
        const _Float16* __restrict__ vt16, float* __restrict__ out) {
    __shared__ _Float16 plds[4][16][32];   // per-wave P re-swizzle scratch (4 KB)

    const int wave = threadIdx.x >> 5;
    const int lane = threadIdx.x & 31;
    const int tile = blockIdx.x * 4 + wave;          // 2048 tiles
    const int n    = tile / (HWS / 16);
    const int s0   = (tile % (HWS / 16)) * 16;
    const int lo   = lane & 15;
    const int hi   = lane >> 4;

    const _Float16* qbase = q16  + (size_t)n * HWS * CCH;
    const _Float16* kbase = k16  + (size_t)n * HWS * CCH;
    const _Float16* vbase = vt16 + (size_t)n * CCH * HWS;

    // Resident Q A-fragments (8 x v16h = 64 VGPRs)
    v16h qf[8];
#pragma unroll
    for (int c0 = 0; c0 < 8; ++c0) {
        const _Float16* p = qbase + (size_t)(s0 + lo) * CCH + c0 * 32 + hi * 16;
        qf[c0] = cat8(*(const v8h*)p, *(const v8h*)(p + 8));
    }

    const v8f vzero = {};
    v8f oacc[16];
#pragma unroll
    for (int j = 0; j < 16; ++j) oacc[j] = vzero;
    float mrow[8], lrow[8];
#pragma unroll
    for (int r = 0; r < 8; ++r) { mrow[r] = -1e30f; lrow[r] = 0.f; }

    for (int t0 = 0; t0 < HWS; t0 += 32) {
        // prefetch next K chunk into cache while this chunk computes
        if (t0 + 32 < HWS)
            __builtin_prefetch(kbase + (size_t)(t0 + 32 + lane) * CCH, 0, 3);

        // ---- S = (Q/sqrtC) * K^T : two 16x16 tiles, double-buffered loads ----
        const _Float16* krow0 = kbase + (size_t)(t0 + lo) * CCH + hi * 16;
        const _Float16* krow1 = kbase + (size_t)(t0 + 16 + lo) * CCH + hi * 16;
        v8f sa = vzero, sb = vzero;
        v16h kb0 = *(const v16h*)krow0;
        v16h kb1 = *(const v16h*)krow1;
#pragma unroll
        for (int c0 = 0; c0 < 8; ++c0) {
            v16h b0 = kb0, b1 = kb1;
            if (c0 < 7) {
                kb0 = *(const v16h*)(krow0 + (c0 + 1) * 32);
                kb1 = *(const v16h*)(krow1 + (c0 + 1) * 32);
            }
            sa = wmma_f16(qf[c0], b0, sa);
            sb = wmma_f16(qf[c0], b1, sb);
        }

        // first V fragment load issued early: latency hides under softmax VALU
        const _Float16* vrow = vbase + (size_t)lo * HWS + t0 + hi * 16;
        v16h vb_next = *(const v16h*)vrow;

        // ---- online softmax (rows 0-7 in lanes 0-15, rows 8-15 in 16-31) ----
#pragma unroll
        for (int r = 0; r < 8; ++r) {
            float a = sa[r], b = sb[r];
            float mx = fmaxf(a, b);
            mx = fmaxf(mx, __shfl_xor(mx, 1));
            mx = fmaxf(mx, __shfl_xor(mx, 2));
            mx = fmaxf(mx, __shfl_xor(mx, 4));
            mx = fmaxf(mx, __shfl_xor(mx, 8));
            float mnew = fmaxf(mrow[r], mx);
            float sc   = __expf(mrow[r] - mnew);
            float p0   = __expf(a - mnew);
            float p1   = __expf(b - mnew);
            float ps   = p0 + p1;
            ps += __shfl_xor(ps, 1);
            ps += __shfl_xor(ps, 2);
            ps += __shfl_xor(ps, 4);
            ps += __shfl_xor(ps, 8);
            lrow[r] = lrow[r] * sc + ps;
            mrow[r] = mnew;
#pragma unroll
            for (int j = 0; j < 16; ++j) oacc[j][r] *= sc;
            int row = r + hi * 8;
            plds[wave][row][lo]      = (_Float16)p0;
            plds[wave][row][16 + lo] = (_Float16)p1;
        }
        asm volatile("s_wait_dscnt 0x0" ::: "memory");

        // ---- P as A-fragment (16x32 fp16) ----
        const _Float16* pp = &plds[wave][lo][hi * 16];
        v16h pfrag = cat8(*(const v8h*)pp, *(const v8h*)(pp + 8));

        // ---- O += P * V : 16 channel tiles, double-buffered loads ----
#pragma unroll
        for (int j = 0; j < 16; ++j) {
            v16h vb = vb_next;
            if (j < 15) vb_next = *(const v16h*)(vrow + (size_t)(j + 1) * 16 * HWS);
            oacc[j] = wmma_f16(pfrag, vb, oacc[j]);
        }
    }

    // ---- epilogue: normalize and scatter to (N,C,H,W) fp32 output ----
#pragma unroll
    for (int r = 0; r < 8; ++r) {
        float inv = 1.0f / lrow[r];
        int srow = s0 + r + hi * 8;
#pragma unroll
        for (int j = 0; j < 16; ++j) {
            out[((size_t)n * CCH + j * 16 + lo) * HWS + srow] = oacc[j][r] * inv;
        }
    }
}

// ---------------------------------------------------------------------------
extern "C" void kernel_launch(void* const* d_in, const int* in_sizes, int n_in,
                              void* d_out, int out_size, void* d_ws, size_t ws_size,
                              hipStream_t stream) {
    (void)in_sizes; (void)n_in; (void)out_size; (void)ws_size;

    const float* x     = (const float*)d_in[0];
    const float* gamma = (const float*)d_in[1];
    const float* beta  = (const float*)d_in[2];
    const float* Wq    = (const float*)d_in[3];
    const float* bq    = (const float*)d_in[4];
    const float* Wk    = (const float*)d_in[5];
    const float* bk    = (const float*)d_in[6];
    const float* Wv    = (const float*)d_in[7];
    const float* bv    = (const float*)d_in[8];

    char* ws = (char*)d_ws;
    const size_t WBYTES = (size_t)CCH * CCH * 2;          // 128 KB each
    const size_t QBYTES = (size_t)NB * HWS * CCH * 2;     // 16 MB each
    _Float16* w16q = (_Float16*)(ws);
    _Float16* w16k = (_Float16*)(ws + WBYTES);
    _Float16* w16v = (_Float16*)(ws + 2 * WBYTES);
    _Float16* q16  = (_Float16*)(ws + 3 * WBYTES);
    _Float16* k16  = (_Float16*)(ws + 3 * WBYTES + QBYTES);
    _Float16* vt16 = (_Float16*)(ws + 3 * WBYTES + 2 * QBYTES);

    cvt_w_kernel<<<CCH * CCH / 256, 256, 0, stream>>>(Wq, Wk, Wv, w16q, w16k, w16v);

    const int tiles = NB * HWS / 16;   // 2048
    lnqkv_kernel<<<tiles / 4, 128, 0, stream>>>(x, gamma, beta,
                                                w16q, bq, w16k, bk, w16v, bv,
                                                q16, k16, vt16);

    attn_kernel<<<tiles / 4, 128, 0, stream>>>(q16, k16, vt16, (float*)d_out);
}
